// Origin_Module_53953379173267
// MI455X (gfx1250) — matrix operand order repaired and checked by reading, not compiled, not run on hardware
//
#include <hip/hip_runtime.h>
#include <hip/hip_bf16.h>

// ---------------------------------------------------------------------------
// 2-layer GraphSAGE forward for MI455X (gfx1250, wave32, WMMA bf16).
// GEMMs stage the per-block weight tile in LDS (ds_load_b128 B-fragments),
// cutting L2 weight traffic 4x and decoupling B reads (DScnt) from A reads
// (LOADcnt) so the scheduler can overlap them with v_wmma.
// ---------------------------------------------------------------------------

typedef __bf16 v16bf __attribute__((ext_vector_type(16)));
typedef __bf16 v8bf  __attribute__((ext_vector_type(8)));
typedef __bf16 v4bf  __attribute__((ext_vector_type(4)));
typedef float  v8f   __attribute__((ext_vector_type(8)));

#define N1c   120000
#define N2c   8192
#define CINc  128
#define CHc   256
#define COUTc 47
#define K2c   512   // layer-2 concat width (256 mean + 256 x_tgt)
#define NP2c  64    // layer-2 padded output cols

static __device__ __forceinline__ void atomAddF(float* p, float v) {
  __hip_atomic_fetch_add(p, v, __ATOMIC_RELAXED, __HIP_MEMORY_SCOPE_AGENT);
}

// --- WMMA fragment loaders -------------------------------------------------
// A (16x32 bf16, row-major, stride lda), global memory:
//   lanes 0-15 : K = kk+0..7 (v0..3) and kk+16..23 (v4..7)
//   lanes 16-31: K = kk+8..15         and kk+24..31
static __device__ __forceinline__ v16bf load_frag_a(const __bf16* __restrict__ base,
                                                    int lda, int m0, int kk, int lane) {
  int row  = m0 + (lane & 15);
  int half = lane >> 4;
  const __bf16* p = base + (size_t)row * lda + kk + half * 8;
  v8bf lo = *(const v8bf*)p;
  v8bf hi = *(const v8bf*)(p + 16);
  v16bf r;
#pragma unroll
  for (int i = 0; i < 8; ++i) { r[i] = lo[i]; r[i + 8] = hi[i]; }
  return r;
}

// B (32x16 bf16) from an LDS tile stored [N][K] row-major (stride ldk):
//   lanes 0-15 : K = kk+0..7 and kk+16..23, lanes 16-31: K = kk+8..15 and kk+24..31
static __device__ __forceinline__ v16bf load_frag_b_lds(const __bf16* sW, int ldk,
                                                        int n0, int kk, int lane) {
  int col  = n0 + (lane & 15);
  int half = lane >> 4;
  const __bf16* p = sW + col * ldk + kk + half * 8;
  v8bf lo = *(const v8bf*)p;
  v8bf hi = *(const v8bf*)(p + 16);
  v16bf r;
#pragma unroll
  for (int i = 0; i < 8; ++i) { r[i] = lo[i]; r[i + 8] = hi[i]; }
  return r;
}

// --- Weight prep: bf16, transposed to [N][K], stacked [Wl;Wr], padded ------
__global__ void prep_w_kernel(const float* __restrict__ W1l, const float* __restrict__ W1r,
                              const float* __restrict__ W2l, const float* __restrict__ W2r,
                              __bf16* __restrict__ Wt1, __bf16* __restrict__ Wt2) {
  int t = blockIdx.x * blockDim.x + threadIdx.x;
  if (t < CHc * CHc) {                       // Wt1: [256 N][256 K]
    int n = t >> 8, k = t & 255;
    float v = (k < CINc) ? W1l[(size_t)k * CHc + n] : W1r[(size_t)(k - CINc) * CHc + n];
    Wt1[(size_t)n * CHc + k] = (__bf16)v;
  } else {
    int t2 = t - CHc * CHc;
    if (t2 < NP2c * K2c) {                   // Wt2: [64 N][512 K], rows >=47 zero
      int n = t2 >> 9, k = t2 & 511;
      float v = 0.f;
      if (n < COUTc)
        v = (k < CHc) ? W2l[(size_t)k * COUTc + n] : W2r[(size_t)(k - CHc) * COUTc + n];
      Wt2[(size_t)n * K2c + k] = (__bf16)v;
    }
  }
}

// --- Layer-1 edge scatter: one wave per edge, 4 f32 channels per lane ------
__global__ void scatter1_kernel(const float* __restrict__ x, const int* __restrict__ src,
                                const int* __restrict__ dst, float* __restrict__ agg,
                                float* __restrict__ cnt, int E) {
  int lane = threadIdx.x & 31;
  int e = blockIdx.x * 8 + (threadIdx.x >> 5);
  if (e >= E) return;
  int s = src[e], d = dst[e];
  float4 v = *(const float4*)(x + (size_t)s * CINc + lane * 4);
  float* o = agg + (size_t)d * CINc + lane * 4;
  atomAddF(o + 0, v.x); atomAddF(o + 1, v.y);
  atomAddF(o + 2, v.z); atomAddF(o + 3, v.w);
  if (lane == 0) atomAddF(cnt + d, 1.0f);
}

// --- Build A1 = bf16([agg/max(cnt,1) || x_tgt])  [N1][256] ----------------
__global__ void build_a1_kernel(const float* __restrict__ agg, const float* __restrict__ cnt,
                                const float* __restrict__ x, __bf16* __restrict__ A1) {
  int idx = blockIdx.x * blockDim.x + threadIdx.x;   // one thread per 4 elems
  if (idx >= N1c * 64) return;
  int row = idx >> 6;
  int c4  = (idx & 63) * 4;
  float4 v;
  if (c4 < CINc) {
    float inv = 1.0f / fmaxf(cnt[row], 1.0f);
    v = *(const float4*)(agg + (size_t)row * CINc + c4);
    v.x *= inv; v.y *= inv; v.z *= inv; v.w *= inv;
  } else {
    v = *(const float4*)(x + (size_t)row * CINc + (c4 - CINc));
  }
  v4bf o; o[0] = (__bf16)v.x; o[1] = (__bf16)v.y; o[2] = (__bf16)v.z; o[3] = (__bf16)v.w;
  *(v4bf*)(A1 + (size_t)row * CHc + c4) = o;
}

// --- GEMM1: h = relu(A1 @ Wt1^T + b1), bf16 out [N1][256] ------------------
// block = 256 thr = 8 waves (4M x 2N); wave tile 16x64; block tile 64x128.
// Weight tile (128 N x 256 K = 64KB) staged in LDS once per block.
__global__ void gemm1_kernel(const __bf16* __restrict__ A1, const __bf16* __restrict__ Wt1,
                             const float* __restrict__ b1, __bf16* __restrict__ h) {
  __shared__ __bf16 sW[128 * CHc];           // 64 KB
  int tid  = threadIdx.x;
  int lane = tid & 31;
  int w    = tid >> 5;

  // Cooperative fill: 128*256 bf16 = 32768 elems; 16 x b128 per thread.
  const __bf16* gw = Wt1 + (size_t)blockIdx.y * 128 * CHc;
#pragma unroll
  for (int i = 0; i < 16; ++i) {
    int off8 = tid * 8 + i * 2048;           // __bf16 units (16B chunks)
    *(v8bf*)(sW + off8) = *(const v8bf*)(gw + off8);
  }
  __syncthreads();

  int m0  = blockIdx.x * 64 + (w & 3) * 16;
  int nl0 = (w >> 2) * 64;                   // N offset inside the LDS tile
  v8f acc[4] = {v8f{0}, v8f{0}, v8f{0}, v8f{0}};

#pragma unroll
  for (int kk = 0; kk < CHc; kk += 32) {
    v16bf a = load_frag_a(A1, CHc, m0, kk, lane);
#pragma unroll
    for (int j = 0; j < 4; ++j) {
      v16bf b = load_frag_b_lds(sW, CHc, nl0 + 16 * j, kk, lane);
      acc[j] = __builtin_amdgcn_wmma_f32_16x16x32_bf16(false, a, false, b,
                                                       (short)0, acc[j], false, false);
    }
  }

  int half = lane >> 4;
#pragma unroll
  for (int j = 0; j < 4; ++j) {
    int col = blockIdx.y * 128 + nl0 + 16 * j + (lane & 15);
    float bias = b1[col];
#pragma unroll
    for (int v = 0; v < 8; ++v) {
      int row = m0 + v + 8 * half;
      float val = acc[j][v] + bias;
      val = val > 0.f ? val : 0.f;
      h[(size_t)row * CHc + col] = (__bf16)val;
    }
  }
}

// --- Layer-2 edge scatter: one wave per edge, 8 bf16 channels per lane -----
__global__ void scatter2_kernel(const __bf16* __restrict__ h, const int* __restrict__ src,
                                const int* __restrict__ dst, float* __restrict__ agg2,
                                float* __restrict__ cnt2, int E) {
  int lane = threadIdx.x & 31;
  int e = blockIdx.x * 8 + (threadIdx.x >> 5);
  if (e >= E) return;
  int s = src[e], d = dst[e];
  v8bf v = *(const v8bf*)(h + (size_t)s * CHc + lane * 8);
  float* o = agg2 + (size_t)d * CHc + lane * 8;
#pragma unroll
  for (int i = 0; i < 8; ++i) atomAddF(o + i, (float)v[i]);
  if (lane == 0) atomAddF(cnt2 + d, 1.0f);
}

// --- Build A2 = bf16([agg2/max(cnt2,1) || h_tgt])  [N2][512] --------------
__global__ void build_a2_kernel(const float* __restrict__ agg2, const float* __restrict__ cnt2,
                                const __bf16* __restrict__ h, __bf16* __restrict__ A2) {
  int idx = blockIdx.x * blockDim.x + threadIdx.x;
  if (idx >= N2c * 128) return;
  int row = idx >> 7;
  int c4  = (idx & 127) * 4;
  if (c4 < CHc) {
    float inv = 1.0f / fmaxf(cnt2[row], 1.0f);
    float4 v = *(const float4*)(agg2 + (size_t)row * CHc + c4);
    v4bf o;
    o[0] = (__bf16)(v.x * inv); o[1] = (__bf16)(v.y * inv);
    o[2] = (__bf16)(v.z * inv); o[3] = (__bf16)(v.w * inv);
    *(v4bf*)(A2 + (size_t)row * K2c + c4) = o;
  } else {
    *(v4bf*)(A2 + (size_t)row * K2c + c4) =
        *(const v4bf*)(h + (size_t)row * CHc + (c4 - CHc));
  }
}

// --- GEMM2: logits = A2 @ Wt2^T + b2, f32 out [N2][64] ---------------------
// block = 8 waves (4M x 2N); wave tile 16x32; block tile 64x64.
// Entire padded weight matrix (64 N x 512 K = 64KB) staged in LDS.
__global__ void gemm2_kernel(const __bf16* __restrict__ A2, const __bf16* __restrict__ Wt2,
                             const float* __restrict__ b2, float* __restrict__ logits) {
  __shared__ __bf16 sW[NP2c * K2c];          // 64 KB
  int tid  = threadIdx.x;
  int lane = tid & 31;
  int w    = tid >> 5;

#pragma unroll
  for (int i = 0; i < 16; ++i) {
    int off8 = tid * 8 + i * 2048;
    *(v8bf*)(sW + off8) = *(const v8bf*)(Wt2 + off8);
  }
  __syncthreads();

  int m0  = blockIdx.x * 64 + (w & 3) * 16;
  int nl0 = (w >> 2) * 32;
  v8f acc[2] = {v8f{0}, v8f{0}};

#pragma unroll
  for (int kk = 0; kk < K2c; kk += 32) {
    v16bf a = load_frag_a(A2, K2c, m0, kk, lane);
#pragma unroll
    for (int j = 0; j < 2; ++j) {
      v16bf b = load_frag_b_lds(sW, K2c, nl0 + 16 * j, kk, lane);
      acc[j] = __builtin_amdgcn_wmma_f32_16x16x32_bf16(false, a, false, b,
                                                       (short)0, acc[j], false, false);
    }
  }

  int half = lane >> 4;
#pragma unroll
  for (int j = 0; j < 2; ++j) {
    int col = nl0 + 16 * j + (lane & 15);
    float bias = (col < COUTc) ? b2[col] : 0.f;
#pragma unroll
    for (int v = 0; v < 8; ++v) {
      int row = m0 + v + 8 * half;
      logits[(size_t)row * NP2c + col] = acc[j][v] + bias;
    }
  }
}

// --- log_softmax over 47 classes, one wave per row -------------------------
__global__ void logsoftmax_kernel(const float* __restrict__ logits, float* __restrict__ out) {
  int lane = threadIdx.x & 31;
  int row  = blockIdx.x * 8 + (threadIdx.x >> 5);
  if (row >= N2c) return;
  const float* lr = logits + (size_t)row * NP2c;
  bool h0 = lane < COUTc;
  bool h1 = (lane + 32) < COUTc;
  float v0 = h0 ? lr[lane] : -3.0e38f;
  float v1 = h1 ? lr[lane + 32] : -3.0e38f;
  float m = fmaxf(v0, v1);
#pragma unroll
  for (int off = 16; off > 0; off >>= 1) m = fmaxf(m, __shfl_xor(m, off, 32));
  float s = (h0 ? __expf(v0 - m) : 0.f) + (h1 ? __expf(v1 - m) : 0.f);
#pragma unroll
  for (int off = 16; off > 0; off >>= 1) s += __shfl_xor(s, off, 32);
  float lse = m + __logf(s);
  float* orow = out + (size_t)row * COUTc;
  if (h0) orow[lane] = v0 - lse;
  if (h1) orow[lane + 32] = v1 - lse;
}

// ---------------------------------------------------------------------------
extern "C" void kernel_launch(void* const* d_in, const int* in_sizes, int n_in,
                              void* d_out, int out_size, void* d_ws, size_t ws_size,
                              hipStream_t stream) {
  const float* x   = (const float*)d_in[0];
  const int* src1  = (const int*)d_in[1];
  const int* dst1  = (const int*)d_in[2];
  const int* src2  = (const int*)d_in[3];
  const int* dst2  = (const int*)d_in[4];
  // d_in[5]=n1, d_in[6]=n2: device scalars; fixed at compile time (120000, 8192)
  const float* W1l = (const float*)d_in[7];
  const float* W1r = (const float*)d_in[8];
  const float* b1  = (const float*)d_in[9];
  const float* W2l = (const float*)d_in[10];
  const float* W2r = (const float*)d_in[11];
  const float* b2  = (const float*)d_in[12];
  float* out = (float*)d_out;
  int E1 = in_sizes[1];
  int E2 = in_sizes[3];

  // Workspace layout (region r0 is reused for layer-2 buffers once agg1 dies).
  char* ws = (char*)d_ws;
  size_t off = 0;
  auto take = [&](size_t bytes) -> char* {
    char* p = ws + off;
    off += (bytes + 255) & ~(size_t)255;
    return p;
  };
  char*   r0   = take((size_t)N1c * CINc * 4);           // 61.44 MB
  float*  agg1 = (float*)r0;
  float*  cnt1 = (float*)take((size_t)N1c * 4);
  __bf16* A1   = (__bf16*)take((size_t)N1c * CHc * 2);   // 61.44 MB
  __bf16* hbuf = (__bf16*)take((size_t)N1c * CHc * 2);   // 61.44 MB
  __bf16* Wt1  = (__bf16*)take((size_t)CHc * CHc * 2);
  __bf16* Wt2  = (__bf16*)take((size_t)NP2c * K2c * 2);
  // layer-2 buffers alias r0 (agg1 is dead after build_a1):
  float*  agg2   = (float*)r0;
  float*  cnt2   = (float*)(r0 + (size_t)N2c * CHc * 4);
  __bf16* A2     = (__bf16*)(r0 + (size_t)N2c * CHc * 4 + 65536);
  float*  logits = (float*)(r0 + (size_t)N2c * CHc * 4 + 65536 + (size_t)N2c * K2c * 2);

  hipMemsetAsync(agg1, 0, (size_t)N1c * CINc * 4, stream);
  hipMemsetAsync(cnt1, 0, (size_t)N1c * 4, stream);
  prep_w_kernel<<<(CHc * CHc + NP2c * K2c + 255) / 256, 256, 0, stream>>>(
      W1l, W1r, W2l, W2r, Wt1, Wt2);
  scatter1_kernel<<<(E1 + 7) / 8, 256, 0, stream>>>(x, src1, dst1, agg1, cnt1, E1);
  build_a1_kernel<<<(N1c * 64 + 255) / 256, 256, 0, stream>>>(agg1, cnt1, x, A1);
  dim3 g1(N1c / 64, 2);
  gemm1_kernel<<<g1, 256, 0, stream>>>(A1, Wt1, b1, hbuf);
  hipMemsetAsync(agg2, 0, (size_t)N2c * CHc * 4, stream);
  hipMemsetAsync(cnt2, 0, (size_t)N2c * 4, stream);
  scatter2_kernel<<<(E2 + 7) / 8, 256, 0, stream>>>(hbuf, src2, dst2, agg2, cnt2, E2);
  build_a2_kernel<<<(N2c * 128 + 255) / 256, 256, 0, stream>>>(agg2, cnt2, hbuf, A2);
  gemm2_kernel<<<N2c / 64, 256, 0, stream>>>(A2, Wt2, b2, logits);
  logsoftmax_kernel<<<N2c / 8, 256, 0, stream>>>(logits, out);
}
